// Decoder_64226940944456
// MI455X (gfx1250) — compile-verified, matrix operand order checked by reading, not run on hardware
//
#include <hip/hip_runtime.h>

typedef __attribute__((ext_vector_type(16))) _Float16 v16h;
typedef __attribute__((ext_vector_type(8)))  _Float16 v8h;
typedef __attribute__((ext_vector_type(8)))  float    v8f;

#define B_    2048
#define L_    64
#define H_    128
#define O_    64
#define T_    512
#define G4H   512   // 4*H

// Branchless fast transcendentals (v_exp_f32 + v_rcp_f32, no exec-mask branches).
__device__ __forceinline__ float frcp_(float x) { return __builtin_amdgcn_rcpf(x); }
__device__ __forceinline__ float sigm_(float x) { return frcp_(1.0f + __expf(-x)); }
__device__ __forceinline__ float tanh_(float x) { return 1.0f - 2.0f * frcp_(1.0f + __expf(2.0f * x)); }

// ---------------- Phase A: step-0 peel (x0 == 0, c0 == 0) ----------------

// h0 = latent @ fc_w^T + fc_b
__global__ void k_h0(const float* __restrict__ latent, const float* __restrict__ fc_w,
                     const float* __restrict__ fc_b, float* __restrict__ h_state) {
  int idx = blockIdx.x * blockDim.x + threadIdx.x;   // over B*H
  int b = idx >> 7, h = idx & (H_ - 1);
  float acc = fc_b[h];
  const float* lp = latent + (size_t)b * L_;
  const float* wp = fc_w + (size_t)h * L_;
#pragma unroll 8
  for (int l = 0; l < L_; ++l) acc = fmaf(lp[l], wp[l], acc);
  h_state[idx] = acc;
}

// gates0 = h0 @ W_hh^T + b_ih + b_hh   (x0 = 0 -> no W_ih term)
__global__ void k_gates0(const float* __restrict__ h_state, const float* __restrict__ W_hh,
                         const float* __restrict__ b_ih, const float* __restrict__ b_hh,
                         float* __restrict__ gates0) {
  int idx = blockIdx.x * blockDim.x + threadIdx.x;   // over B*4H
  int b = idx >> 9, r = idx & (G4H - 1);
  float acc = b_ih[r] + b_hh[r];
  const float* hp = h_state + (size_t)b * H_;
  const float* wp = W_hh + (size_t)r * H_;
#pragma unroll 8
  for (int k = 0; k < H_; ++k) acc = fmaf(hp[k], wp[k], acc);
  gates0[idx] = acc;
}

// c1 = sig(i)*tanh(g) ; h1 = sig(o)*tanh(c1)   (c0 = 0 -> forget term drops)
__global__ void k_step0(const float* __restrict__ gates0,
                        float* __restrict__ h_state, float* __restrict__ c_state) {
  int idx = blockIdx.x * blockDim.x + threadIdx.x;   // over B*H
  int b = idx >> 7, j = idx & (H_ - 1);
  const float* g = gates0 + (size_t)b * G4H;
  float gi = g[j], gg = g[2 * H_ + j], go = g[3 * H_ + j];
  float c = sigm_(gi) * tanh_(gg);
  c_state[idx] = c;
  h_state[idx] = sigm_(go) * tanh_(c);
}

// ---------------- Phase B: persistent recurrence over T steps ----------------
// One block = 16 batch rows, 8 waves. W_comb = W_ih + W_hh resident in VGPRs
// as WMMA B-fragments (wave w owns gate N-tiles 4w..4w+3); gate bias resident
// as C-operand splats. Wo fragments live in LDS. c in registers, h in LDS (f16).
// Loop body: one A-fragment load of h_t feeds BOTH gates_{t+1} and out_t.

__launch_bounds__(256, 1)
__global__ void k_lstm(const float* __restrict__ W_ih, const float* __restrict__ W_hh,
                       const float* __restrict__ b_ih, const float* __restrict__ b_hh,
                       const float* __restrict__ Wo,   const float* __restrict__ bo,
                       const float* __restrict__ h_state, const float* __restrict__ c_state,
                       float* __restrict__ out) {
  __shared__ float gates_lds[16][G4H + 1];
  __shared__ __align__(16) _Float16 h_lds[16][H_ + 8];
  __shared__ __align__(16) _Float16 wo_lds[4][4][32][16];   // [outTile][kt][lane][elem]

  const int tid    = threadIdx.x;
  const int wave   = tid >> 5;
  const int lane   = tid & 31;
  const int laneHi = (lane >> 4) & 1;
  const int lm     = lane & 15;
  const int bbase  = blockIdx.x * 16;

  // --- load W_comb fragments into registers (ISA 16-bit B layout) ---
  v16h wf[4][4];
  float gbias[4];
#pragma unroll
  for (int nt = 0; nt < 4; ++nt) {
    const int grow = (wave * 4 + nt) * 16 + lm;           // gate row 0..511
    gbias[nt] = b_ih[grow] + b_hh[grow];                  // per-column bias for C init
    const float* wi = W_ih + (size_t)grow * H_;
    const float* wh = W_hh + (size_t)grow * H_;
#pragma unroll
    for (int kt = 0; kt < 4; ++kt) {
#pragma unroll
      for (int e = 0; e < 16; ++e) {
        const int k = kt * 32 + ((e < 8) ? e : (e + 8)) + laneHi * 8;
        wf[nt][kt][e] = (_Float16)(wi[k] + wh[k]);
      }
    }
  }
  // --- output-projection fragments into LDS (Wo row = out column) ---
  if (wave < 4) {
    const int orow = wave * 16 + lm;                      // 0..63
    const float* wo = Wo + (size_t)orow * H_;
#pragma unroll
    for (int kt = 0; kt < 4; ++kt)
#pragma unroll
      for (int e = 0; e < 16; ++e) {
        const int k = kt * 32 + ((e < 8) ? e : (e + 8)) + laneHi * 8;
        wo_lds[wave][kt][lane][e] = (_Float16)wo[k];
      }
  }
  const float obias = bo[(wave & 3) * 16 + lm];

  // --- initial state: h1 into LDS (f16), c1 into registers ---
  const int er  = tid >> 4;          // row 0..15
  const int ejb = (tid & 15) * 8;    // 8 consecutive H columns
  float creg[8];
#pragma unroll
  for (int q = 0; q < 8; ++q) {
    const size_t gix = (size_t)(bbase + er) * H_ + (ejb + q);
    creg[q] = c_state[gix];
    h_lds[er][ejb + q] = (_Float16)h_state[gix];
  }
  __syncthreads();

#pragma unroll 1
  for (int t = 0; t < T_; ++t) {
    // ---- one A-fragment load of h_t, shared by gates_{t+1} and out_t ----
    v16h a[4];
#pragma unroll
    for (int kt = 0; kt < 4; ++kt) {
      const _Float16* hp = &h_lds[lm][kt * 32 + laneHi * 8];
      v8h lo = *(const v8h*)(hp);
      v8h hi = *(const v8h*)(hp + 16);
      a[kt] = __builtin_shufflevector(lo, hi, 0,1,2,3,4,5,6,7,8,9,10,11,12,13,14,15);
    }

    // ---- gate GEMM: gates_{t+1} = h_t @ W_comb^T + b_comb ----
#pragma unroll
    for (int nt = 0; nt < 4; ++nt) {
      const float gb = gbias[nt];
      v8f acc = {gb, gb, gb, gb, gb, gb, gb, gb};   // bias in C operand
#pragma unroll
      for (int kt = 0; kt < 4; ++kt)
        acc = __builtin_amdgcn_wmma_f32_16x16x32_f16(false, a[kt], false, wf[nt][kt],
                                                     (short)0, acc, false, false);
      const int col = (wave * 4 + nt) * 16 + lm;
#pragma unroll
      for (int r = 0; r < 8; ++r)
        gates_lds[r + laneHi * 8][col] = acc[r];
    }

    // ---- output projection: out[:, t, :] = h_t @ Wo^T + bo (waves 0-3) ----
    if (wave < 4) {
      v8f acc = {};
#pragma unroll
      for (int kt = 0; kt < 4; ++kt) {
        const _Float16* wp = &wo_lds[wave][kt][lane][0];
        v8h lo = *(const v8h*)(wp);
        v8h hi = *(const v8h*)(wp + 8);
        v16h bfrag = __builtin_shufflevector(lo, hi, 0,1,2,3,4,5,6,7,8,9,10,11,12,13,14,15);
        acc = __builtin_amdgcn_wmma_f32_16x16x32_f16(false, a[kt], false, bfrag,
                                                     (short)0, acc, false, false);
      }
      const int col = wave * 16 + lm;
#pragma unroll
      for (int r = 0; r < 8; ++r) {
        const int brow = bbase + r + laneHi * 8;
        out[((size_t)brow * T_ + t) * O_ + col] = acc[r] + obias;
      }
    }
    __syncthreads();

    // ---- elementwise LSTM cell update: (gates_{t+1}, c_t) -> (h_{t+1}, c_{t+1}) ----
#pragma unroll
    for (int q = 0; q < 8; ++q) {
      const int j = ejb + q;
      const float gi = gates_lds[er][j];
      const float gf = gates_lds[er][H_ + j];
      const float gg = gates_lds[er][2 * H_ + j];
      const float go = gates_lds[er][3 * H_ + j];
      const float c  = sigm_(gf) * creg[q] + sigm_(gi) * tanh_(gg);
      creg[q] = c;
      h_lds[er][j] = (_Float16)(sigm_(go) * tanh_(c));
    }
    __syncthreads();
  }
}

// ---------------- host-side launch ----------------

extern "C" void kernel_launch(void* const* d_in, const int* in_sizes, int n_in,
                              void* d_out, int out_size, void* d_ws, size_t ws_size,
                              hipStream_t stream) {
  const float* latent = (const float*)d_in[0];
  const float* fc_w   = (const float*)d_in[1];
  const float* fc_b   = (const float*)d_in[2];
  const float* W_ih   = (const float*)d_in[3];
  const float* W_hh   = (const float*)d_in[4];
  const float* b_ih   = (const float*)d_in[5];
  const float* b_hh   = (const float*)d_in[6];
  const float* Wo     = (const float*)d_in[7];
  const float* bo     = (const float*)d_in[8];
  float* out = (float*)d_out;

  float* h_state = (float*)d_ws;                       // B*H
  float* c_state = h_state + (size_t)B_ * H_;          // B*H
  float* gates0  = c_state + (size_t)B_ * H_;          // B*4H

  k_h0    <<<(B_ * H_)  / 256, 256, 0, stream>>>(latent, fc_w, fc_b, h_state);
  k_gates0<<<(B_ * G4H) / 256, 256, 0, stream>>>(h_state, W_hh, b_ih, b_hh, gates0);
  k_step0 <<<(B_ * H_)  / 256, 256, 0, stream>>>(gates0, h_state, c_state);
  k_lstm  <<<B_ / 16,          256, 0, stream>>>(W_ih, W_hh, b_ih, b_hh, Wo, bo,
                                                 h_state, c_state, out);
}
